// PointNetBackbone_62234076119236
// MI455X (gfx1250) — matrix operand order repaired and checked by reading, not compiled, not run on hardware
//
#include <hip/hip_runtime.h>

typedef __attribute__((ext_vector_type(16))) _Float16 v16h;
typedef __attribute__((ext_vector_type(8)))  _Float16 h8;
typedef __attribute__((ext_vector_type(8)))  float    v8f;

#define B_  2
#define N_  16384

static inline int rup(int x, int a) { return (x + a - 1) / a * a; }
static inline int cdiv(int a, int b) { return (a + b - 1) / b; }

// ---------------------------------------------------------------------------
// split points (B,N,5) -> xyz (B,N,3), feats (B,N,2)
// ---------------------------------------------------------------------------
__global__ void split_pts(const float* __restrict__ pts, float* __restrict__ xyz,
                          float* __restrict__ feats, int total) {
  int t = blockIdx.x * blockDim.x + threadIdx.x;
  if (t >= total) return;                       // total = B*N
  const float* p = pts + (size_t)t * 5;
  xyz[(size_t)t * 3 + 0] = p[0];
  xyz[(size_t)t * 3 + 1] = p[1];
  xyz[(size_t)t * 3 + 2] = p[2];
  feats[(size_t)t * 2 + 0] = p[3];
  feats[(size_t)t * 2 + 1] = p[4];
}

// ---------------------------------------------------------------------------
// Farthest point sampling: one block per batch, LDS tree argmax per iteration
// ---------------------------------------------------------------------------
__global__ void fps_kernel(const float* __restrict__ xyz, int* __restrict__ outIdx,
                           float* __restrict__ dist, int N, int npoint) {
  const int b = blockIdx.x;
  const int T = blockDim.x;        // 256
  const int tid = threadIdx.x;
  __shared__ float sv[256];
  __shared__ int   si[256];
  __shared__ int   s_last;
  const float* X = xyz + (size_t)b * N * 3;
  float* D = dist + (size_t)b * N;
  for (int i = tid; i < N; i += T) D[i] = 1e10f;
  if (tid == 0) { outIdx[b * npoint] = 0; s_last = 0; }
  __syncthreads();
  for (int j = 1; j < npoint; ++j) {
    int last = s_last;
    float lx = X[last * 3 + 0], ly = X[last * 3 + 1], lz = X[last * 3 + 2];
    float best = -1.0f; int bi = 0;
    for (int i = tid; i < N; i += T) {
      float dx = X[i * 3 + 0] - lx;
      float dy = X[i * 3 + 1] - ly;
      float dz = X[i * 3 + 2] - lz;
      float d = dx * dx + dy * dy + dz * dz;
      float old = D[i];
      float nd = d < old ? d : old;
      if (nd != old) D[i] = nd;
      if (nd > best) { best = nd; bi = i; }
    }
    sv[tid] = best; si[tid] = bi;
    __syncthreads();
    for (int s = T >> 1; s > 0; s >>= 1) {
      if (tid < s) {
        if (sv[tid + s] > sv[tid]) { sv[tid] = sv[tid + s]; si[tid] = si[tid + s]; }
      }
      __syncthreads();
    }
    if (tid == 0) { s_last = si[0]; outIdx[b * npoint + j] = si[0]; }
    __syncthreads();
  }
}

// ---------------------------------------------------------------------------
// gather xyz by fps indices
// ---------------------------------------------------------------------------
__global__ void gather_xyz(const float* __restrict__ xyz, const int* __restrict__ idx,
                           float* __restrict__ out, int Nsrc, int np, int total) {
  int t = blockIdx.x * blockDim.x + threadIdx.x;
  if (t >= total) return;                       // total = B*np*3
  int c = t % 3;
  int p = (t / 3) % np;
  int b = t / (3 * np);
  out[t] = xyz[((size_t)b * Nsrc + idx[b * np + p]) * 3 + c];
}

// ---------------------------------------------------------------------------
// ball query: first nsample indices (ascending) within radius, padded w/ first
// ---------------------------------------------------------------------------
__global__ void ball_query(const float* __restrict__ xyz, const float* __restrict__ nxyz,
                           int* __restrict__ gi, int Nsrc, int np, float r2, int k,
                           int total) {
  int t = blockIdx.x * blockDim.x + threadIdx.x;
  if (t >= total) return;                       // total = B*np
  int b = t / np;
  float px = nxyz[(size_t)t * 3 + 0];
  float py = nxyz[(size_t)t * 3 + 1];
  float pz = nxyz[(size_t)t * 3 + 2];
  const float* X = xyz + (size_t)b * Nsrc * 3;
  int* out = gi + (size_t)t * k;
  int cnt = 0, first = -1;
  for (int i = 0; i < Nsrc && cnt < k; ++i) {
    float dx = X[i * 3 + 0] - px;
    float dy = X[i * 3 + 1] - py;
    float dz = X[i * 3 + 2] - pz;
    float d = dx * dx + dy * dy + dz * dz;
    if (d < r2) { if (first < 0) first = i; out[cnt++] = i; }
  }
  if (first < 0) first = 0;
  for (; cnt < k; ++cnt) out[cnt] = first;
}

// ---------------------------------------------------------------------------
// build grouped activation: concat(gxyz - new_xyz, feats[gi]) -> f16 (M x Kpad)
// ---------------------------------------------------------------------------
__global__ void sa_group(const float* __restrict__ xyz, const float* __restrict__ feats,
                         const float* __restrict__ nxyz, const int* __restrict__ gi,
                         _Float16* __restrict__ act, int Nsrc, int np, int k, int Cf,
                         int Kpad, int total) {
  int t = blockIdx.x * blockDim.x + threadIdx.x;
  if (t >= total) return;                       // total = B*np*k*Kpad
  int c = t % Kpad;
  int m = t / Kpad;
  int s = m % k;
  int p = (m / k) % np;
  int b = m / (k * np);
  int g = gi[((size_t)b * np + p) * k + s];
  float v = 0.0f;
  if (c < 3) {
    v = xyz[((size_t)b * Nsrc + g) * 3 + c] - nxyz[((size_t)b * np + p) * 3 + c];
  } else if (c < 3 + Cf) {
    v = feats[((size_t)b * Nsrc + g) * Cf + (c - 3)];
  }
  act[t] = (_Float16)v;
}

// ---------------------------------------------------------------------------
// weight convert f32 (cin x cout) -> transposed padded f16 Wh[n*Kpad + k]
// ---------------------------------------------------------------------------
__global__ void convert_w(const float* __restrict__ W, _Float16* __restrict__ Wh,
                          int cin, int cout, int Kpad, int total) {
  int t = blockIdx.x * blockDim.x + threadIdx.x;
  if (t >= total) return;                       // total = Npad*Kpad
  int kk = t % Kpad;
  int n = t / Kpad;
  float v = (kk < cin && n < cout) ? W[(size_t)kk * cout + n] : 0.0f;
  Wh[t] = (_Float16)v;
}

// ---------------------------------------------------------------------------
// WMMA GEMM + bias + ReLU.  2x2 register-blocked: one 32x32 output per wave32
// (4 WMMAs per 128B of fragment loads -> 16 flop/byte vs 8 for 1x1).
//   A  : (M x K) row-major f16, K multiple of 32; M multiple of 32 in all uses
//   Wh : transposed weights, Wh[n*K + k]; outStride multiple of 32
//   out: f16 with stride outStride (zero-filled pad cols)  OR  f32 tight
// ---------------------------------------------------------------------------
__device__ __forceinline__ void store_one(_Float16* outH, float* outF, size_t idx,
                                          float acc, float b, bool valid) {
  float v = valid ? fmaxf(acc + b, 0.0f) : 0.0f;
  if (outH) outH[idx] = (_Float16)v;
  else      outF[idx] = v;
}

__global__ void gemm_wmma(const _Float16* __restrict__ A, const _Float16* __restrict__ Wh,
                          const float* __restrict__ bias, _Float16* __restrict__ outH,
                          float* __restrict__ outF, int M, int K, int cout, int outStride) {
  const int lane = threadIdx.x & 31;
  const int wave = blockIdx.x * 8 + (threadIdx.x >> 5);
  const int nT = outStride >> 5;                // 32-wide column blocks
  const int mT = (M + 31) >> 5;
  const int mTile = wave / nT;
  const int nTile = wave - mTile * nT;
  if (mTile >= mT) return;                      // wave-uniform: EXEC stays full
  const int half = lane >> 4;
  const int l15 = lane & 15;
  int m0 = mTile * 32 + l15;                    // M % 32 == 0 in all uses
  int m1 = m0 + 16;
  if (m0 >= M) m0 = M - 1;                      // safety clamp (never hit here)
  if (m1 >= M) m1 = M - 1;
  const int col0 = nTile * 32 + l15;
  const int col1 = col0 + 16;
  const _Float16* arow0 = A + (size_t)m0 * K;
  const _Float16* arow1 = A + (size_t)m1 * K;
  const _Float16* bcol0 = Wh + (size_t)col0 * K;
  const _Float16* bcol1 = Wh + (size_t)col1 * K;
  v8f acc00 = {}, acc01 = {}, acc10 = {}, acc11 = {};
  for (int k0 = 0; k0 < K; k0 += 32) {
    __builtin_prefetch(arow0 + k0 + 256, 0, 1);   // global_prefetch_b8
    __builtin_prefetch(arow1 + k0 + 256, 0, 1);
    v16h a0 = *(const v16h*)(arow0 + k0 + 16 * half);
    v16h a1 = *(const v16h*)(arow1 + k0 + 16 * half);
    v16h b0 = *(const v16h*)(bcol0 + k0 + 16 * half);
    v16h b1 = *(const v16h*)(bcol1 + k0 + 16 * half);
    acc00 = __builtin_amdgcn_wmma_f32_16x16x32_f16(false, a0, false, b0,
                                                   (short)0, acc00, false, false);
    acc01 = __builtin_amdgcn_wmma_f32_16x16x32_f16(false, a0, false, b1,
                                                   (short)0, acc01, false, false);
    acc10 = __builtin_amdgcn_wmma_f32_16x16x32_f16(false, a1, false, b0,
                                                   (short)0, acc10, false, false);
    acc11 = __builtin_amdgcn_wmma_f32_16x16x32_f16(false, a1, false, b1,
                                                   (short)0, acc11, false, false);
  }
  const bool v0 = col0 < cout;
  const bool v1 = col1 < cout;
  const float b0v = v0 ? bias[col0] : 0.0f;
  const float b1v = v1 ? bias[col1] : 0.0f;
#pragma unroll
  for (int r = 0; r < 8; ++r) {
    int r0 = mTile * 32 + r + 8 * half;
    int r1 = r0 + 16;
    if (r0 < M) {
      store_one(outH, outF, (size_t)r0 * outStride + col0, acc00[r], b0v, v0);
      store_one(outH, outF, (size_t)r0 * outStride + col1, acc01[r], b1v, v1);
    }
    if (r1 < M) {
      store_one(outH, outF, (size_t)r1 * outStride + col0, acc10[r], b0v, v0);
      store_one(outH, outF, (size_t)r1 * outStride + col1, acc11[r], b1v, v1);
    }
  }
}

// ---------------------------------------------------------------------------
// max over the k grouped samples -> f32 feature tensor (concat along channels)
// ---------------------------------------------------------------------------
__global__ void maxpool(const _Float16* __restrict__ act, float* __restrict__ out,
                        int np, int k, int cout, int actStride, int Ctot, int coff,
                        int total) {
  int t = blockIdx.x * blockDim.x + threadIdx.x;
  if (t >= total) return;                       // total = B*np*cout
  int c = t % cout;
  int p = (t / cout) % np;
  int b = t / (cout * np);
  size_t base = ((size_t)(b * np + p) * k) * actStride + c;
  float m = -1e30f;
  for (int s = 0; s < k; ++s) m = fmaxf(m, (float)act[base + (size_t)s * actStride]);
  out[((size_t)b * np + p) * Ctot + coff + c] = m;
}

// ---------------------------------------------------------------------------
// 3-NN search + inverse-distance weights
// ---------------------------------------------------------------------------
__global__ void three_nn(const float* __restrict__ txyz, const float* __restrict__ sxyz,
                         int* __restrict__ nnIdx, float* __restrict__ nnW,
                         int Nt, int Ns, int total) {
  int t = blockIdx.x * blockDim.x + threadIdx.x;
  if (t >= total) return;                       // total = B*Nt
  int b = t / Nt;
  float px = txyz[(size_t)t * 3 + 0];
  float py = txyz[(size_t)t * 3 + 1];
  float pz = txyz[(size_t)t * 3 + 2];
  const float* S = sxyz + (size_t)b * Ns * 3;
  float d0 = 1e30f, d1 = 1e30f, d2 = 1e30f;
  int i0 = 0, i1 = 0, i2 = 0;
  for (int i = 0; i < Ns; ++i) {
    float dx = S[i * 3 + 0] - px;
    float dy = S[i * 3 + 1] - py;
    float dz = S[i * 3 + 2] - pz;
    float d = dx * dx + dy * dy + dz * dz;
    if (d < d0)      { d2 = d1; i2 = i1; d1 = d0; i1 = i0; d0 = d; i0 = i; }
    else if (d < d1) { d2 = d1; i2 = i1; d1 = d;  i1 = i; }
    else if (d < d2) { d2 = d;  i2 = i; }
  }
  float w0 = 1.0f / (d0 + 1e-8f);
  float w1 = 1.0f / (d1 + 1e-8f);
  float w2 = 1.0f / (d2 + 1e-8f);
  float ws = w0 + w1 + w2;
  nnIdx[(size_t)t * 3 + 0] = i0;
  nnIdx[(size_t)t * 3 + 1] = i1;
  nnIdx[(size_t)t * 3 + 2] = i2;
  nnW[(size_t)t * 3 + 0] = w0 / ws;
  nnW[(size_t)t * 3 + 1] = w1 / ws;
  nnW[(size_t)t * 3 + 2] = w2 / ws;
}

// ---------------------------------------------------------------------------
// FP concat: [interp(src_feats), tgt_feats] -> f16 (M x Kpad)
// ---------------------------------------------------------------------------
__global__ void fp_concat(const float* __restrict__ srcF, const float* __restrict__ tgtF,
                          const int* __restrict__ nnIdx, const float* __restrict__ nnW,
                          _Float16* __restrict__ act, int Nt, int Ns, int Csrc, int Ctgt,
                          int Kpad, int total) {
  int t = blockIdx.x * blockDim.x + threadIdx.x;
  if (t >= total) return;                       // total = B*Nt*Kpad
  int c = t % Kpad;
  int m = t / Kpad;
  int p = m % Nt;
  int b = m / Nt;
  float v = 0.0f;
  if (c < Csrc) {
    const int* I = nnIdx + (size_t)m * 3;
    const float* W = nnW + (size_t)m * 3;
    const float* S = srcF + (size_t)b * Ns * Csrc;
    v = W[0] * S[(size_t)I[0] * Csrc + c] +
        W[1] * S[(size_t)I[1] * Csrc + c] +
        W[2] * S[(size_t)I[2] * Csrc + c];
  } else if (c < Csrc + Ctgt) {
    v = tgtF[((size_t)b * Nt + p) * Ctgt + (c - Csrc)];
  }
  act[t] = (_Float16)v;
}

// ===========================================================================
// host orchestration
// ===========================================================================
static void launch_convw(const float* W, _Float16* Wh, int cin, int cout,
                         hipStream_t st) {
  int Kpad = rup(cin, 32), Npad = rup(cout, 32);
  int tot = Kpad * Npad;
  convert_w<<<cdiv(tot, 256), 256, 0, st>>>(W, Wh, cin, cout, Kpad, tot);
}

static void launch_gemm(const _Float16* A, const _Float16* Wh, const float* bias,
                        _Float16* oh, float* of, int M, int K, int cout, int outStride,
                        hipStream_t st) {
  long waves = (long)cdiv(M, 32) * (outStride / 32);
  gemm_wmma<<<(int)((waves + 7) / 8), 256, 0, st>>>(A, Wh, bias, oh, of, M, K, cout,
                                                    outStride);
}

extern "C" void kernel_launch(void* const* d_in, const int* in_sizes, int n_in,
                              void* d_out, int out_size, void* d_ws, size_t ws_size,
                              hipStream_t stream) {
  (void)in_sizes; (void)n_in; (void)out_size; (void)ws_size;

  // -------- configuration tables --------
  static const int   NP[4]      = {4096, 1024, 256, 64};
  static const float R2T[4][2]  = {{0.01f, 0.25f}, {0.25f, 1.0f}, {1.0f, 4.0f}, {4.0f, 16.0f}};
  static const int   KS[2]      = {16, 32};
  static const int   CF[4]      = {2, 96, 256, 512};   // input feature channels per level
  static const int   CTOT[4]    = {96, 256, 512, 1024};
  static const int   COUT[4][2][3] = {
      {{16, 16, 32},  {32, 32, 64}},
      {{64, 64, 128}, {64, 96, 128}},
      {{128, 196, 256}, {128, 196, 256}},
      {{256, 256, 512}, {256, 384, 512}}};
  static const int FPC[4][3] = {{1536, 512, 512}, {768, 512, 512},
                                {608, 256, 256},  {258, 128, 128}};

  const float* points = (const float*)d_in[0];
  auto saW = [&](int l, int s, int j) { return (const float*)d_in[1 + ((l * 2 + s) * 3 + j) * 2]; };
  auto saB = [&](int l, int s, int j) { return (const float*)d_in[2 + ((l * 2 + s) * 3 + j) * 2]; };
  auto fpW = [&](int i, int j) { return (const float*)d_in[49 + (i * 2 + j) * 2]; };
  auto fpB = [&](int i, int j) { return (const float*)d_in[50 + (i * 2 + j) * 2]; };

  // -------- workspace bump allocator --------
  char* wp = (char*)d_ws;
  auto alloc = [&](size_t bytes) -> void* {
    void* r = (void*)wp;
    wp += (bytes + 255) & ~(size_t)255;
    return r;
  };
  float* xyz0   = (float*)alloc((size_t)B_ * N_ * 3 * 4);
  float* feats0 = (float*)alloc((size_t)B_ * N_ * 2 * 4);
  float* dist   = (float*)alloc((size_t)B_ * N_ * 4);
  int*   idxBuf = (int*)alloc((size_t)B_ * 4096 * 4);
  float* xyzL[4]; float* featsL[4];
  for (int l = 0; l < 4; ++l) {
    xyzL[l]   = (float*)alloc((size_t)B_ * NP[l] * 3 * 4);
    featsL[l] = (float*)alloc((size_t)B_ * NP[l] * CTOT[l] * 4);
  }
  int*   gi    = (int*)alloc((size_t)B_ * 4096 * 32 * 4);
  int*   nnIdx = (int*)alloc((size_t)B_ * N_ * 3 * 4);
  float* nnW   = (float*)alloc((size_t)B_ * N_ * 3 * 4);
  float* fpf[3];
  fpf[0] = (float*)alloc((size_t)B_ * 256 * 512 * 4);
  fpf[1] = (float*)alloc((size_t)B_ * 1024 * 512 * 4);
  fpf[2] = (float*)alloc((size_t)B_ * 4096 * 256 * 4);
  _Float16* wbuf = (_Float16*)alloc((size_t)1536 * 512 * 2);
  const size_t ACT_ELEMS = (size_t)16777216;    // max: 262144 rows x 64 cols
  _Float16* actA = (_Float16*)alloc(ACT_ELEMS * 2);
  _Float16* actB = (_Float16*)alloc(ACT_ELEMS * 2);

  // -------- split input --------
  {
    int tot = B_ * N_;
    split_pts<<<cdiv(tot, 256), 256, 0, stream>>>(points, xyz0, feats0, tot);
  }

  // -------- SA levels --------
  for (int l = 0; l < 4; ++l) {
    const float* sx = (l == 0) ? xyz0 : xyzL[l - 1];
    const float* sf = (l == 0) ? feats0 : featsL[l - 1];
    int Ns = (l == 0) ? N_ : NP[l - 1];
    int np = NP[l];

    fps_kernel<<<B_, 256, 0, stream>>>(sx, idxBuf, dist, Ns, np);
    {
      int tot = B_ * np * 3;
      gather_xyz<<<cdiv(tot, 256), 256, 0, stream>>>(sx, idxBuf, xyzL[l], Ns, np, tot);
    }

    int coff = 0;
    for (int s = 0; s < 2; ++s) {
      int k = KS[s];
      {
        int tot = B_ * np;
        ball_query<<<cdiv(tot, 256), 256, 0, stream>>>(sx, xyzL[l], gi, Ns, np,
                                                       R2T[l][s], k, tot);
      }
      int cin0 = 3 + CF[l];
      int Kp0 = rup(cin0, 32);
      int M = B_ * np * k;
      {
        int tot = M * Kp0;
        sa_group<<<cdiv(tot, 256), 256, 0, stream>>>(sx, sf, xyzL[l], gi, actA, Ns, np,
                                                     k, CF[l], Kp0, tot);
      }
      _Float16* bufs[2] = {actA, actB};
      int ci = 0;
      int cin = cin0;
      for (int j = 0; j < 3; ++j) {
        int cout = COUT[l][s][j];
        launch_convw(saW(l, s, j), wbuf, cin, cout, stream);
        launch_gemm(bufs[ci], wbuf, saB(l, s, j), bufs[ci ^ 1], nullptr, M,
                    rup(cin, 32), cout, rup(cout, 32), stream);
        ci ^= 1;
        cin = cout;
      }
      {
        int tot = B_ * np * cin;
        maxpool<<<cdiv(tot, 256), 256, 0, stream>>>(bufs[ci], featsL[l], np, k, cin,
                                                    rup(cin, 32), CTOT[l], coff, tot);
      }
      coff += cin;
    }
  }

  // -------- FP modules --------
  const float* tgtXyz[4] = {xyzL[2], xyzL[1], xyzL[0], xyz0};
  const float* srcXyz[4] = {xyzL[3], xyzL[2], xyzL[1], xyzL[0]};
  const float* tgtF[4]   = {featsL[2], featsL[1], featsL[0], feats0};
  const float* srcF[4]   = {featsL[3], fpf[0], fpf[1], fpf[2]};
  const int    tgtN[4]   = {256, 1024, 4096, 16384};
  const int    srcN[4]   = {64, 256, 1024, 4096};
  const int    Ctgt[4]   = {512, 256, 96, 2};
  const int    Csrc[4]   = {1024, 512, 512, 256};
  float*       outBuf[4] = {fpf[0], fpf[1], fpf[2], (float*)d_out};

  for (int i = 0; i < 4; ++i) {
    int Nt = tgtN[i], Ns = srcN[i];
    int M = B_ * Nt;
    {
      int tot = B_ * Nt;
      three_nn<<<cdiv(tot, 256), 256, 0, stream>>>(tgtXyz[i], srcXyz[i], nnIdx, nnW,
                                                   Nt, Ns, tot);
    }
    int C0 = Csrc[i] + Ctgt[i];
    int Kp0 = rup(C0, 32);
    {
      int tot = M * Kp0;
      fp_concat<<<cdiv(tot, 256), 256, 0, stream>>>(srcF[i], tgtF[i], nnIdx, nnW, actA,
                                                    Nt, Ns, Csrc[i], Ctgt[i], Kp0, tot);
    }
    int c1 = FPC[i][1], c2 = FPC[i][2];
    launch_convw(fpW(i, 0), wbuf, C0, c1, stream);
    launch_gemm(actA, wbuf, fpB(i, 0), actB, nullptr, M, Kp0, c1, rup(c1, 32), stream);
    launch_convw(fpW(i, 1), wbuf, c1, c2, stream);
    // last layer of each module writes tight f32 (module 3 -> d_out)
    launch_gemm(actB, wbuf, fpB(i, 1), nullptr, outBuf[i], M, rup(c1, 32), c2, c2,
                stream);
  }
}
